// LSTM_GCN_61529701483364
// MI455X (gfx1250) — matrix operand order repaired and checked by reading, not compiled, not run on hardware
//
#include <hip/hip_runtime.h>
#include <hip/hip_bf16.h>

// ---------------------------------------------------------------------------
// Problem constants (match the reference)
// ---------------------------------------------------------------------------
#define Bb   8
#define Tt   64
#define Nn   512
#define Hh   128
#define G1c  128
#define G2c  64
#define Ec   8192
#define Ss   (Bb * Nn)      // 4096 sequences
#define G4   (4 * Hh)       // 512 gate columns

typedef __attribute__((ext_vector_type(16))) _Float16 v16h;
typedef __attribute__((ext_vector_type(8)))  _Float16 v8h;
typedef __attribute__((ext_vector_type(8)))  float    v8f;

// ---------------------------------------------------------------------------
// CDNA5 async global->LDS copy (CDNA5 ISA, GV mode).
// LDS address taken from the vdst VGPR; tracked by ASYNCcnt.
// ---------------------------------------------------------------------------
__device__ __forceinline__ void async_copy_b128(const void* gaddr, void* lds) {
  asm volatile("global_load_async_to_lds_b128 %0, %1, off"
               :
               : "v"((unsigned)(uintptr_t)lds), "v"(gaddr)
               : "memory");
}
__device__ __forceinline__ void wait_async0() {
  asm volatile("s_wait_asynccnt 0" ::: "memory");
}

// ---------------------------------------------------------------------------
// Fragment helpers (layouts per CDNA5 ISA 7.12.2)
// A (16x32 f16): lane L -> row M = L&15; halves j<8 -> K = kb + (L>>4)*8 + j,
//                j>=8 -> K = kb + 16 + (L>>4)*8 + (j-8)
// B (32x16 f16): pre-swizzled in memory as [frag][lane][16 halves]
// C/D (16x16 f32): vgpr r, lane L -> row M = (L>>4)*8 + r, col N = L&15
// ---------------------------------------------------------------------------
__device__ __forceinline__ v16h load_a_frag(const _Float16* base, int rowStride,
                                            int kbase, int lane) {
  const int M  = lane & 15;
  const int hi = lane >> 4;
  const _Float16* p = base + (size_t)M * rowStride + kbase + hi * 8;
  union { v16h v; v8h h[2]; } u;
  u.h[0] = *(const v8h*)(p);
  u.h[1] = *(const v8h*)(p + 16);
  return u.v;
}

__device__ __forceinline__ v16h load_b_frag(const _Float16* Wf, int fragIdx, int lane) {
  return *(const v16h*)(Wf + ((size_t)fragIdx * 32 + lane) * 16);
}

__device__ __forceinline__ float sigm_f(float x) {
  return 1.0f / (1.0f + __expf(-x));
}
__device__ __forceinline__ float tanh_f(float x) {
  float e = __expf(2.0f * x);
  return (e - 1.0f) / (e + 1.0f);
}

// ---------------------------------------------------------------------------
// Weight packing: B[k][n] = (k<Ka ? Wa[n*Ka+k] : Wb[n*Kb+(k-Ka)])  -> f16 frags
// frag f = kt*(N/16)+nt ; within frag: (lane, j) -> k = kt*32+(j>>3)*16+(lane>>4)*8+(j&7),
//                                                  n = nt*16+(lane&15)
// ---------------------------------------------------------------------------
__global__ void pack_b_frags_kernel(const float* __restrict__ Wa, int Ka,
                                    const float* __restrict__ Wb, int Kb,
                                    int N, int K, _Float16* __restrict__ out) {
  int idx = blockIdx.x * blockDim.x + threadIdx.x;
  if (idx >= K * N) return;
  int j    = idx & 15;
  int lane = (idx >> 4) & 31;
  int f    = idx >> 9;
  int ntiles = N >> 4;
  int kt = f / ntiles;
  int nt = f % ntiles;
  int k = kt * 32 + (j >> 3) * 16 + (lane >> 4) * 8 + (j & 7);
  int n = nt * 16 + (lane & 15);
  float v = (k < Ka) ? Wa[(size_t)n * Ka + k] : Wb[(size_t)n * Kb + (k - Ka)];
  out[idx] = (_Float16)v;
}

__global__ void vec_add_kernel(const float* __restrict__ a, const float* __restrict__ b,
                               float* __restrict__ o, int n) {
  int i = blockIdx.x * blockDim.x + threadIdx.x;
  if (i < n) o[i] = a[i] + b[i];
}

__global__ void vec_copy_kernel(const float* __restrict__ a, float* __restrict__ o, int n) {
  int i = blockIdx.x * blockDim.x + threadIdx.x;
  if (i < n) o[i] = a[i];
}

__global__ void zero_f32_kernel(float* __restrict__ p, int n) {
  int i = blockIdx.x * blockDim.x + threadIdx.x;
  if (i < n) p[i] = 0.0f;
}

// ---------------------------------------------------------------------------
// LSTM layer: one workgroup handles 32 sequences through all T steps.
// 8 waves: wave w -> mtile = w>>2 (16 seqs), q = w&3 (32 hidden units).
// Each wave keeps all 4 gates for its (16 seq x 32 unit) patch in 8 f32
// accumulators -> cell update fully in registers. h_t lives in LDS as f16.
// Template param L0: layer0 (scalar input, K=128) vs layers 1/2 (K=256 with
// the input projection fused in as K-tiles 0..3) -- compile-time specialized
// so the kt loop fully unrolls and B fragments stay register-resident.
// ---------------------------------------------------------------------------
template <int L0>
__global__ __launch_bounds__(256, 1)
void lstm_layer_kernel(const _Float16* __restrict__ Wfrag,  // [ktiles*32 frags][32][16]
                       const float* __restrict__ wb,        // [512] b_ih + b_hh
                       const float* __restrict__ wi0,       // [512] w_ih col (layer0)
                       const float* __restrict__ x,         // [B,T,N] (layer0)
                       const _Float16* __restrict__ hin,    // [S,T,H] (layers 1,2)
                       _Float16* __restrict__ hout) {       // [S,T,H]
  __shared__ __align__(32) _Float16 aH[32 * Hh];   // h_{t-1} tile, f16
  __shared__ __align__(32) _Float16 aIn[32 * Hh];  // prev-layer h_t tile, f16
  __shared__ float xv[32];                         // scalar input (layer0)

  const int tid   = threadIdx.x;
  const int lane  = tid & 31;
  const int wv    = tid >> 5;
  const int mbase = (wv >> 2) * 16;
  const int q     = wv & 3;
  const int sBase = blockIdx.x * 32;

  // per-lane constants over time: biases and layer0 input weights
  float bias[4][2], wic[4][2];
  #pragma unroll
  for (int gi = 0; gi < 4; ++gi)
    #pragma unroll
    for (int uu = 0; uu < 2; ++uu) {
      int col = gi * Hh + q * 32 + uu * 16 + (lane & 15);
      bias[gi][uu] = wb[col];
      wic[gi][uu]  = L0 ? wi0[col] : 0.0f;
    }

  // init h=0 in LDS, load input for t=0
  for (int i = tid; i < 32 * Hh; i += 256) aH[i] = (_Float16)0.0f;
  if (L0) {
    if (tid < 32) {
      int s = sBase + tid, b = s >> 9, n = s & 511;
      xv[tid] = x[(size_t)b * Tt * Nn + n];
    }
  } else {
    int off = tid * 16, s = off >> 7, k = off & 127;
    const _Float16* src = &hin[((size_t)(sBase + s) * Tt + 0) * Hh + k];
    async_copy_b128(src, &aIn[off]);
    async_copy_b128(src + 8, &aIn[off + 8]);
  }

  v8f c0, c1;
  #pragma unroll
  for (int i = 0; i < 8; ++i) { c0[i] = 0.0f; c1[i] = 0.0f; }

  constexpr int ktiles = L0 ? 4 : 8;

  for (int t = 0; t < Tt; ++t) {
    if (!L0) wait_async0();   // aIn(t) async copy complete (per wave)
    __syncthreads();          // aH(t-1), aIn(t)/xv(t) visible to all waves

    v8f acc[4][2];
    v8f zz;
    #pragma unroll
    for (int i = 0; i < 8; ++i) zz[i] = 0.0f;
    #pragma unroll
    for (int gi = 0; gi < 4; ++gi)
      #pragma unroll
      for (int uu = 0; uu < 2; ++uu) acc[gi][uu] = zz;

    #pragma unroll
    for (int kt = 0; kt < ktiles; ++kt) {
      const _Float16* asrc;
      int kb;
      if (L0)          { asrc = aH;  kb = kt * 32; }
      else if (kt < 4) { asrc = aIn; kb = kt * 32; }
      else             { asrc = aH;  kb = (kt - 4) * 32; }
      v16h af = load_a_frag(asrc + mbase * Hh, Hh, kb, lane);
      #pragma unroll
      for (int gi = 0; gi < 4; ++gi)
        #pragma unroll
        for (int uu = 0; uu < 2; ++uu) {
          int nt = gi * 8 + q * 2 + uu;
          v16h bf = load_b_frag(Wfrag, kt * 32 + nt, lane);
          acc[gi][uu] = __builtin_amdgcn_wmma_f32_16x16x32_f16(
              false, af, false, bf, (short)0, acc[gi][uu], false, false);
        }
    }

    float xrow[8];
    if (L0) {
      #pragma unroll
      for (int r = 0; r < 8; ++r) xrow[r] = xv[mbase + (lane >> 4) * 8 + r];
    }

    __syncthreads();  // all waves done reading aH before overwrite

    #pragma unroll
    for (int uu = 0; uu < 2; ++uu) {
      v8f& cc = uu ? c1 : c0;
      #pragma unroll
      for (int r = 0; r < 8; ++r) {
        float gI = acc[0][uu][r] + bias[0][uu];
        float gF = acc[1][uu][r] + bias[1][uu];
        float gG = acc[2][uu][r] + bias[2][uu];
        float gO = acc[3][uu][r] + bias[3][uu];
        if (L0) {
          float xr = xrow[r];
          gI += xr * wic[0][uu]; gF += xr * wic[1][uu];
          gG += xr * wic[2][uu]; gO += xr * wic[3][uu];
        }
        float cv = sigm_f(gF) * cc[r] + sigm_f(gI) * tanh_f(gG);
        cc[r] = cv;
        float hv = sigm_f(gO) * tanh_f(cv);
        int seq = mbase + (lane >> 4) * 8 + r;
        int col = q * 32 + uu * 16 + (lane & 15);
        aH[seq * Hh + col] = (_Float16)hv;
        hout[((size_t)(sBase + seq) * Tt + t) * Hh + col] = (_Float16)hv;
      }
    }

    if (t + 1 < Tt) {
      if (L0) {
        if (tid < 32) {
          int s = sBase + tid, b = s >> 9, n = s & 511;
          xv[tid] = x[(size_t)b * Tt * Nn + (size_t)(t + 1) * Nn + n];
        }
      } else {
        // async prefetch of next timestep's input tile, completed before the
        // next iteration's WMMA work; completion enforced by wait_async0().
        int off = tid * 16, s = off >> 7, k = off & 127;
        const _Float16* src = &hin[((size_t)(sBase + s) * Tt + (t + 1)) * Hh + k];
        async_copy_b128(src, &aIn[off]);
        async_copy_b128(src + 8, &aIn[off + 8]);
      }
    }
  }
}

// ---------------------------------------------------------------------------
// Generic WMMA GEMM: C[M,N] = A_f16[M,K] @ Bfrag ; one 16x16 tile per wave
// ---------------------------------------------------------------------------
__global__ __launch_bounds__(256, 2)
void wmma_gemm_kernel(const _Float16* __restrict__ A, size_t rowStride,
                      const _Float16* __restrict__ Bf, float* __restrict__ C,
                      int M, int N, int K) {
  const int lane = threadIdx.x & 31;
  const int wv   = threadIdx.x >> 5;
  const int tileId = blockIdx.x * 8 + wv;
  const int ntiles = N >> 4;
  if (tileId >= (M >> 4) * ntiles) return;
  const int mtile = tileId / ntiles;
  const int nt    = tileId % ntiles;

  v8f acc;
  #pragma unroll
  for (int i = 0; i < 8; ++i) acc[i] = 0.0f;

  const _Float16* Abase = A + (size_t)(mtile * 16) * rowStride;
  for (int kt = 0; kt < (K >> 5); ++kt) {
    v16h af = load_a_frag(Abase, (int)rowStride, kt * 32, lane);
    v16h bf = load_b_frag(Bf, kt * ntiles + nt, lane);
    acc = __builtin_amdgcn_wmma_f32_16x16x32_f16(false, af, false, bf,
                                                 (short)0, acc, false, false);
  }
  #pragma unroll
  for (int r = 0; r < 8; ++r) {
    int row = mtile * 16 + (lane >> 4) * 8 + r;
    int col = nt * 16 + (lane & 15);
    C[(size_t)row * N + col] = acc[r];
  }
}

// ---------------------------------------------------------------------------
// GCN support kernels
// ---------------------------------------------------------------------------
__global__ void degree_kernel(const int* __restrict__ ei, float* __restrict__ deg) {
  int idx = blockIdx.x * blockDim.x + threadIdx.x;
  if (idx >= Ec + Nn) return;
  int d = (idx < Ec) ? ei[Ec + idx] : (idx - Ec);
  atomicAdd(&deg[d], 1.0f);
}

__global__ void gcn_scatter_kernel(const float* __restrict__ xl,
                                   const int* __restrict__ ei,
                                   const float* __restrict__ deg,
                                   float* __restrict__ agg, int C) {
  int idx = blockIdx.x * blockDim.x + threadIdx.x;
  int total = (Ec + Nn) * C;
  if (idx >= total) return;
  int e = idx / C, c = idx % C;
  int s, d;
  if (e < Ec) { s = ei[e]; d = ei[Ec + e]; } else { s = d = e - Ec; }
  float norm = rsqrtf(deg[s] * deg[d]);
  #pragma unroll
  for (int b = 0; b < Bb; ++b) {
    atomicAdd(&agg[((size_t)b * Nn + d) * C + c],
              xl[((size_t)b * Nn + s) * C + c] * norm);
  }
}

__global__ void gcn_finalize_f16_kernel(const float* __restrict__ agg,
                                        const float* __restrict__ bias,
                                        _Float16* __restrict__ out, int C, int total) {
  int idx = blockIdx.x * blockDim.x + threadIdx.x;
  if (idx >= total) return;
  float v = agg[idx] + bias[idx % C];
  out[idx] = (_Float16)(v > 0.0f ? v : 0.0f);
}

__global__ void pool_kernel(const float* __restrict__ agg2,
                            const float* __restrict__ b2,
                            float* __restrict__ pooled) {
  int idx = blockIdx.x * blockDim.x + threadIdx.x;
  if (idx >= Bb * G2c) return;
  int b = idx / G2c, c = idx % G2c;
  float bias = b2[c], sum = 0.0f;
  for (int n = 0; n < Nn; ++n) {
    float v = agg2[((size_t)b * Nn + n) * G2c + c] + bias;
    sum += (v > 0.0f ? v : 0.0f);
  }
  pooled[idx] = sum * (1.0f / Nn);
}

__global__ void cls_kernel(const float* __restrict__ pooled,
                           const float* __restrict__ cls_w,
                           const float* __restrict__ cls_b,
                           float* __restrict__ out) {
  int b = threadIdx.x;
  if (b >= Bb) return;
  float acc = cls_b[0];
  for (int c = 0; c < G2c; ++c) acc += pooled[b * G2c + c] * cls_w[c];
  out[b] = acc;
}

// ---------------------------------------------------------------------------
// Host-side orchestration
// ---------------------------------------------------------------------------
extern "C" void kernel_launch(void* const* d_in, const int* in_sizes, int n_in,
                              void* d_out, int out_size, void* d_ws, size_t ws_size,
                              hipStream_t stream) {
  (void)in_sizes; (void)n_in; (void)out_size; (void)ws_size;
  const float* x     = (const float*)d_in[0];
  const int*   ei    = (const int*)d_in[1];
  const float* wih[3] = {(const float*)d_in[2], (const float*)d_in[6],  (const float*)d_in[10]};
  const float* whh[3] = {(const float*)d_in[3], (const float*)d_in[7],  (const float*)d_in[11]};
  const float* bih[3] = {(const float*)d_in[4], (const float*)d_in[8],  (const float*)d_in[12]};
  const float* bhh[3] = {(const float*)d_in[5], (const float*)d_in[9],  (const float*)d_in[13]};
  const float* g1w = (const float*)d_in[14];
  const float* g1b = (const float*)d_in[15];
  const float* g2w = (const float*)d_in[16];
  const float* g2b = (const float*)d_in[17];
  const float* clw = (const float*)d_in[18];
  const float* clb = (const float*)d_in[19];
  float* outp = (float*)d_out;

  char* ws = (char*)d_ws;
  size_t off = 0;
  auto alloc = [&](size_t bytes) -> void* {
    void* p = ws + off;
    off = (off + bytes + 255) & ~(size_t)255;
    return p;
  };

  _Float16* W0f = (_Float16*)alloc((size_t)4 * 32 * 512 * 2);  // K=128, N=512
  _Float16* W1f = (_Float16*)alloc((size_t)8 * 32 * 512 * 2);  // K=256, N=512
  _Float16* W2f = (_Float16*)alloc((size_t)8 * 32 * 512 * 2);
  float* wb0 = (float*)alloc(G4 * 4);
  float* wb1 = (float*)alloc(G4 * 4);
  float* wb2 = (float*)alloc(G4 * 4);
  float* wi0 = (float*)alloc(G4 * 4);
  _Float16* G1f = (_Float16*)alloc((size_t)4 * 8 * 512 * 2);   // K=128, N=128
  _Float16* G2f = (_Float16*)alloc((size_t)4 * 4 * 512 * 2);   // K=128, N=64
  _Float16* hA  = (_Float16*)alloc((size_t)Ss * Tt * Hh * 2);  // 64 MB
  _Float16* hB  = (_Float16*)alloc((size_t)Ss * Tt * Hh * 2);  // 64 MB
  float* xl1  = (float*)alloc((size_t)Ss * G1c * 4);
  float* agg1 = (float*)alloc((size_t)Ss * G1c * 4);
  _Float16* h1h = (_Float16*)alloc((size_t)Ss * G1c * 2);
  float* xl2  = (float*)alloc((size_t)Ss * G2c * 4);
  float* agg2 = (float*)alloc((size_t)Ss * G2c * 4);
  float* deg  = (float*)alloc(Nn * 4);
  float* pooled = (float*)alloc(Bb * G2c * 4);

  // ---- weight prep ----
  pack_b_frags_kernel<<<(128 * 512 + 255) / 256, 256, 0, stream>>>(wih[0], 0, whh[0], 128, 512, 128, W0f);
  pack_b_frags_kernel<<<(256 * 512 + 255) / 256, 256, 0, stream>>>(wih[1], 128, whh[1], 128, 512, 256, W1f);
  pack_b_frags_kernel<<<(256 * 512 + 255) / 256, 256, 0, stream>>>(wih[2], 128, whh[2], 128, 512, 256, W2f);
  pack_b_frags_kernel<<<(128 * 128 + 255) / 256, 256, 0, stream>>>(g1w, 0, g1w, 128, 128, 128, G1f);
  pack_b_frags_kernel<<<(128 * 64 + 255) / 256, 256, 0, stream>>>(g2w, 0, g2w, 128, 64, 128, G2f);
  vec_add_kernel<<<2, 256, 0, stream>>>(bih[0], bhh[0], wb0, G4);
  vec_add_kernel<<<2, 256, 0, stream>>>(bih[1], bhh[1], wb1, G4);
  vec_add_kernel<<<2, 256, 0, stream>>>(bih[2], bhh[2], wb2, G4);
  vec_copy_kernel<<<2, 256, 0, stream>>>(wih[0], wi0, G4);  // in_size=1 => column vector

  // ---- zero accumulation buffers (re-done every call: graph-replay safe) ----
  zero_f32_kernel<<<(Ss * G1c + 255) / 256, 256, 0, stream>>>(agg1, Ss * G1c);
  zero_f32_kernel<<<(Ss * G2c + 255) / 256, 256, 0, stream>>>(agg2, Ss * G2c);
  zero_f32_kernel<<<2, 256, 0, stream>>>(deg, Nn);
  degree_kernel<<<(Ec + Nn + 255) / 256, 256, 0, stream>>>(ei, deg);

  // ---- 3-layer LSTM (WMMA recurrent GEMMs, async-LDS activation prefetch) ----
  lstm_layer_kernel<1><<<Ss / 32, 256, 0, stream>>>(W0f, wb0, wi0, x, hB, hA);
  lstm_layer_kernel<0><<<Ss / 32, 256, 0, stream>>>(W1f, wb1, wi0, x, hA, hB);
  lstm_layer_kernel<0><<<Ss / 32, 256, 0, stream>>>(W2f, wb2, wi0, x, hB, hA);

  // ---- GCN layer 1 ----
  wmma_gemm_kernel<<<(Ss / 16) * (G1c / 16) / 8, 256, 0, stream>>>(
      hA + (size_t)(Tt - 1) * Hh, (size_t)Tt * Hh, G1f, xl1, Ss, G1c, Hh);
  gcn_scatter_kernel<<<((Ec + Nn) * G1c + 255) / 256, 256, 0, stream>>>(xl1, ei, deg, agg1, G1c);
  gcn_finalize_f16_kernel<<<(Ss * G1c + 255) / 256, 256, 0, stream>>>(agg1, g1b, h1h, G1c, Ss * G1c);

  // ---- GCN layer 2 ----
  wmma_gemm_kernel<<<(Ss / 16) * (G2c / 16) / 8, 256, 0, stream>>>(
      h1h, (size_t)G1c, G2f, xl2, Ss, G2c, G1c);
  gcn_scatter_kernel<<<((Ec + Nn) * G2c + 255) / 256, 256, 0, stream>>>(xl2, ei, deg, agg2, G2c);

  // ---- pool + classifier ----
  pool_kernel<<<(Bb * G2c + 255) / 256, 256, 0, stream>>>(agg2, g2b, pooled);
  cls_kernel<<<1, 32, 0, stream>>>(pooled, clw, clb, outp);
}
